// GeneLink_5351529251391
// MI455X (gfx1250) — matrix operand order repaired and checked by reading, not compiled, not run on hardware
//
#include <hip/hip_runtime.h>
#include <hip/hip_bf16.h>

// ---------------- problem constants (match reference) ----------------
#define N_NODES 50000
#define N_EDGES 800000
#define IN_CH   128
#define HID     16
#define HEADS   4
#define L1C     (HEADS*HID)   // 64
#define OUT_CH  32
#define NEG_SLOPE 0.2f
#define EPS_      1e-16f

typedef __attribute__((ext_vector_type(16))) _Float16 v16h;
typedef __attribute__((ext_vector_type(8)))  float    v8f;

// ---------------- small helpers ----------------
__device__ __forceinline__ float leaky(float v) {
    return v > 0.0f ? v : v * NEG_SLOPE;
}

// float atomic max via sign-split int/uint ordering trick.
// Works with slots initialized to -inf (0xFF800000).
__device__ __forceinline__ void atomicMaxFloat(float* addr, float val) {
    if (val >= 0.0f) atomicMax((int*)addr, __float_as_int(val));
    else             atomicMin((unsigned int*)addr, __float_as_uint(val));
}

__global__ void fill_kernel(float* p, float v, int n) {
    int i = blockIdx.x * blockDim.x + threadIdx.x;
    if (i < n) p[i] = v;
}

// ---------------- WMMA GEMM: Y[M,NCOL] = X[M,K] @ W[K,NCOL] ----------------
// One 16-row tile per block; wave w owns output col-tile w (NT waves).
// W is converted to f16 and pre-swizzled into WMMA B-fragment layout in LDS:
//   ldsW[((kk*NT + colTile)*32 + lane)*16 + j]
// so each lane reads its whole v16h B fragment as one contiguous 32-byte
// LDS access (2x ds_load_b128) instead of 16 strided u16 loads.
// A fragments come from global per the ISA 16-bit A layout
// (lanes 0-15: K 0..7 / 16..23; lanes 16-31: K 8..15 / 24..31).
template<int K, int NCOL, int NT>
__global__ void __launch_bounds__(32 * NT)
gemm_wmma(const float* __restrict__ X, const float* __restrict__ W,
          float* __restrict__ Y) {
    __shared__ __align__(32) _Float16 ldsW[K * NCOL];

    // swizzling fill: element (k, col) -> B-fragment slot
    for (int e = threadIdx.x; e < K * NCOL; e += 32 * NT) {
        const int k    = e / NCOL;
        const int col  = e - k * NCOL;
        const int kk   = k >> 5;
        const int krem = k & 31;
        const int t    = col >> 4;
        const int lane = (col & 15) + ((krem & 8) ? 16 : 0);
        const int j    = (krem & 7) + ((krem >> 4) << 3);
        ldsW[(((kk * NT) + t) * 32 + lane) * 16 + j] = (_Float16)W[e];
    }
    __syncthreads();

    const int lane  = threadIdx.x & 31;
    const int wv    = threadIdx.x >> 5;          // col tile index
    const int tileM = blockIdx.x;                // 16 rows per block
    const int mrow  = tileM * 16 + (lane & 15);
    const int hi    = lane >> 4;                 // half-wave select
    const int col   = wv * 16 + (lane & 15);

    const float* __restrict__ xr = X + (size_t)mrow * K;

    v8f acc = {};
#pragma unroll
    for (int kk = 0; kk < K / 32; ++kk) {
        // A fragment (16x32 f16) from global, f32 -> f16
        v16h a;
        const int kb = kk * 32 + hi * 8;
#pragma unroll
        for (int j = 0; j < 8; ++j) {
            a[j]     = (_Float16)xr[kb + j];
            a[8 + j] = (_Float16)xr[kb + 16 + j];
        }
        // B fragment (32x16 f16): one contiguous 32B LDS read per lane
        const v16h b = *(const v16h*)&ldsW[(((kk * NT) + wv) * 32 + lane) * 16];

        acc = __builtin_amdgcn_wmma_f32_16x16x32_f16(
            false, a, false, b, (short)0, acc, false, false);
    }

    // C/D layout: lanes 0-15 rows 0..7, lanes 16-31 rows 8..15; col = lane&15
    const int rb = hi * 8;
#pragma unroll
    for (int r = 0; r < 8; ++r)
        Y[(size_t)(tileM * 16 + rb + r) * NCOL + col] = acc[r];
}

// ---------------- per-node attention logits ----------------
// as[n,h] = dot(h[n,h,:], a_src[h,:]);  ad likewise.
template<int H, int C>
__global__ void attn_prep(const float* __restrict__ hfeat,
                          const float* __restrict__ a_src,
                          const float* __restrict__ a_dst,
                          float* __restrict__ as, float* __restrict__ ad) {
    int n = blockIdx.x * blockDim.x + threadIdx.x;
    if (n >= N_NODES) return;
    const float* hp = hfeat + (size_t)n * H * C;
#pragma unroll
    for (int h = 0; h < H; ++h) {
        float s = 0.0f, d = 0.0f;
#pragma unroll
        for (int c = 0; c < C; ++c) {
            float v = hp[h * C + c];
            s += v * a_src[h * C + c];
            d += v * a_dst[h * C + c];
        }
        as[n * H + h] = s;
        ad[n * H + h] = d;
    }
}

// ---------------- edge passes ----------------
template<int H>
__global__ void edge_max(const int* __restrict__ src, const int* __restrict__ dst,
                         const float* __restrict__ as, const float* __restrict__ ad,
                         float* __restrict__ m) {
    int e = blockIdx.x * blockDim.x + threadIdx.x;
    if (e >= N_EDGES) return;
    int s = src[e], d = dst[e];
#pragma unroll
    for (int h = 0; h < H; ++h) {
        float v = leaky(as[s * H + h] + ad[d * H + h]);
        atomicMaxFloat(&m[d * H + h], v);
    }
}

template<int H>
__global__ void edge_expsum(const int* __restrict__ src, const int* __restrict__ dst,
                            const float* __restrict__ as, const float* __restrict__ ad,
                            const float* __restrict__ m, float* __restrict__ den) {
    int e = blockIdx.x * blockDim.x + threadIdx.x;
    if (e >= N_EDGES) return;
    int s = src[e], d = dst[e];
#pragma unroll
    for (int h = 0; h < H; ++h) {
        float v = leaky(as[s * H + h] + ad[d * H + h]);
        atomicAdd(&den[d * H + h], __expf(v - m[d * H + h]));
    }
}

template<int H, int C>
__global__ void edge_scatter(const int* __restrict__ src, const int* __restrict__ dst,
                             const float* __restrict__ as, const float* __restrict__ ad,
                             const float* __restrict__ m, const float* __restrict__ den,
                             const float* __restrict__ hfeat, float* __restrict__ accum) {
    int e = blockIdx.x * blockDim.x + threadIdx.x;
    if (e >= N_EDGES) return;
    int s = src[e], d = dst[e];
    float alpha[H];
#pragma unroll
    for (int h = 0; h < H; ++h) {
        float v = leaky(as[s * H + h] + ad[d * H + h]);
        alpha[h] = __expf(v - m[d * H + h]) / (den[d * H + h] + EPS_);
    }
    const float* hs = hfeat + (size_t)s * H * C;
    float*      acc = accum + (size_t)d * H * C;
#pragma unroll
    for (int h = 0; h < H; ++h)
#pragma unroll
        for (int c = 0; c < C; ++c)
            atomicAdd(&acc[h * C + c], hs[h * C + c] * alpha[h]);
}

// ---------------- epilogues ----------------
// layer1: in-place  acc = elu(acc + b1)
__global__ void finalize_elu(float* __restrict__ acc, const float* __restrict__ b) {
    int i = blockIdx.x * blockDim.x + threadIdx.x;
    if (i >= N_NODES * L1C) return;
    float v = acc[i] + b[i % L1C];
    acc[i] = v > 0.0f ? v : (__expf(v) - 1.0f);
}

// layer2: out = acc + b2   (heads=1, mean over 1 head == identity)
__global__ void finalize_out(const float* __restrict__ acc, const float* __restrict__ b,
                             float* __restrict__ out) {
    int i = blockIdx.x * blockDim.x + threadIdx.x;
    if (i >= N_NODES * OUT_CH) return;
    out[i] = acc[i] + b[i % OUT_CH];
}

// ---------------- host-side launcher ----------------
extern "C" void kernel_launch(void* const* d_in, const int* in_sizes, int n_in,
                              void* d_out, int out_size, void* d_ws, size_t ws_size,
                              hipStream_t stream) {
    const float* x      = (const float*)d_in[0];
    const int*   ei     = (const int*)  d_in[1];
    const float* W1     = (const float*)d_in[2];
    const float* a_src1 = (const float*)d_in[3];
    const float* a_dst1 = (const float*)d_in[4];
    const float* b1     = (const float*)d_in[5];
    const float* W2     = (const float*)d_in[6];
    const float* a_src2 = (const float*)d_in[7];
    const float* a_dst2 = (const float*)d_in[8];
    const float* b2     = (const float*)d_in[9];
    float* out = (float*)d_out;

    const int* src = ei;
    const int* dst = ei + N_EDGES;

    // workspace carve-up (floats)
    float* ws  = (float*)d_ws;
    float* h1  = ws;                               // N*64  layer1 linear output
    float* as1 = h1  + (size_t)N_NODES * L1C;      // N*4
    float* ad1 = as1 + (size_t)N_NODES * HEADS;    // N*4
    float* m1  = ad1 + (size_t)N_NODES * HEADS;    // N*4
    float* dn1 = m1  + (size_t)N_NODES * HEADS;    // N*4
    float* ac1 = dn1 + (size_t)N_NODES * HEADS;    // N*64 accum -> elu in place
    float* h2  = ac1 + (size_t)N_NODES * L1C;      // N*32  layer2 linear output
    float* as2 = h2  + (size_t)N_NODES * OUT_CH;   // N
    float* ad2 = as2 + (size_t)N_NODES;            // N
    float* m2  = ad2 + (size_t)N_NODES;            // N
    float* dn2 = m2  + (size_t)N_NODES;            // N
    float* ac2 = dn2 + (size_t)N_NODES;            // N*32

    const int TB = 256;
    auto blocks = [](int n, int tb) { return (n + tb - 1) / tb; };
    const int egrid = blocks(N_EDGES, TB);
    const int ngrid = blocks(N_NODES, TB);

    // ---------------- layer 1 ----------------
    fill_kernel<<<blocks(N_NODES * HEADS, TB), TB, 0, stream>>>(m1, -INFINITY, N_NODES * HEADS);
    fill_kernel<<<blocks(N_NODES * HEADS, TB), TB, 0, stream>>>(dn1, 0.0f, N_NODES * HEADS);
    fill_kernel<<<blocks(N_NODES * L1C, TB), TB, 0, stream>>>(ac1, 0.0f, N_NODES * L1C);

    gemm_wmma<IN_CH, L1C, 4><<<N_NODES / 16, 32 * 4, 0, stream>>>(x, W1, h1);
    attn_prep<HEADS, HID><<<ngrid, TB, 0, stream>>>(h1, a_src1, a_dst1, as1, ad1);

    edge_max<HEADS><<<egrid, TB, 0, stream>>>(src, dst, as1, ad1, m1);
    edge_expsum<HEADS><<<egrid, TB, 0, stream>>>(src, dst, as1, ad1, m1, dn1);
    edge_scatter<HEADS, HID><<<egrid, TB, 0, stream>>>(src, dst, as1, ad1, m1, dn1, h1, ac1);

    finalize_elu<<<blocks(N_NODES * L1C, TB), TB, 0, stream>>>(ac1, b1);  // ac1 := elu(ac1+b1)

    // ---------------- layer 2 ----------------
    fill_kernel<<<ngrid, TB, 0, stream>>>(m2, -INFINITY, N_NODES);
    fill_kernel<<<ngrid, TB, 0, stream>>>(dn2, 0.0f, N_NODES);
    fill_kernel<<<blocks(N_NODES * OUT_CH, TB), TB, 0, stream>>>(ac2, 0.0f, N_NODES * OUT_CH);

    gemm_wmma<L1C, OUT_CH, 2><<<N_NODES / 16, 32 * 2, 0, stream>>>(ac1, W2, h2);
    attn_prep<1, OUT_CH><<<ngrid, TB, 0, stream>>>(h2, a_src2, a_dst2, as2, ad2);

    edge_max<1><<<egrid, TB, 0, stream>>>(src, dst, as2, ad2, m2);
    edge_expsum<1><<<egrid, TB, 0, stream>>>(src, dst, as2, ad2, m2, dn2);
    edge_scatter<1, OUT_CH><<<egrid, TB, 0, stream>>>(src, dst, as2, ad2, m2, dn2, h2, ac2);

    finalize_out<<<blocks(N_NODES * OUT_CH, TB), TB, 0, stream>>>(ac2, b2, out);
}
